// AttentionMechanism_86036784873880
// MI455X (gfx1250) — compile-verified, matrix operand order checked by reading, not run on hardware
//
#include <hip/hip_runtime.h>
#include <hip/hip_bf16.h>
#include <cstdint>

// ---------------------------------------------------------------------------
// MI455X (gfx1250) attention block, all matrix math on v_wmma_f32_16x16x32_bf16:
//   qkv = x @ w_qkv          (fp32 -> bf16 on the fly, f32 accum)
//   flash-attention, causal additive mask (+1 allowed / -10000 masked)
//   out = o @ w_out          (f32 output)
// Global->LDS staging uses gfx1250 async loads (ASYNCcnt-tracked
// global_load_async_to_lds_b128), guarded by __has_builtin.
// ---------------------------------------------------------------------------

typedef __bf16 bf16_t;
typedef bf16_t v16bf __attribute__((ext_vector_type(16)));
typedef bf16_t v8bf  __attribute__((ext_vector_type(8)));
typedef bf16_t v4bf  __attribute__((ext_vector_type(4)));
typedef float  v8f   __attribute__((ext_vector_type(8)));

#define WMMA_BF16(a, b, c) \
  __builtin_amdgcn_wmma_f32_16x16x32_bf16(false, (a), false, (b), (short)0, (c), false, false)

#if __has_builtin(__builtin_amdgcn_global_load_async_to_lds_b128)
#define HAS_ASYNC_LDS 1
#else
#define HAS_ASYNC_LDS 0
#endif

// Builtin parameter types (from hipcc diagnostic): 4 x i32 GCC vector,
// AS1 (global, printed as __device__) source, AS3 (LDS) destination.
typedef int v4i __attribute__((vector_size(16)));
typedef __attribute__((address_space(1))) v4i as1_v4i;
typedef __attribute__((address_space(3))) v4i as3_v4i;

// 16-byte global->LDS copy: async (ASYNCcnt) when available, else via VGPRs.
__device__ __forceinline__ void copy16_g2l(const bf16_t* g, bf16_t* l) {
#if HAS_ASYNC_LDS
  __builtin_amdgcn_global_load_async_to_lds_b128(
      (as1_v4i*)(uintptr_t)g, (as3_v4i*)(uint32_t)(uintptr_t)l, 0, 0);
#else
  *(v8bf*)l = *(const v8bf*)g;
#endif
}

__device__ __forceinline__ void async_stage_wait() {
#if HAS_ASYNC_LDS
#if __has_builtin(__builtin_amdgcn_s_wait_asynccnt)
  __builtin_amdgcn_s_wait_asynccnt(0);
#else
  asm volatile("s_wait_asynccnt 0x0" ::: "memory");
#endif
#endif
}

// 16x32 bf16 fragment load (A-layout; B via transposed tiles).
// ISA 7.12.2: lane L -> row=L%16, half=L/16; v0..3 = K(half*8..+7),
// v4..7 = K(16+half*8..+7). Rows must be 16B aligned.
__device__ __forceinline__ v16bf load_frag16x32(const bf16_t* tile, int rowstride) {
  const int lane = threadIdx.x & 31;
  const bf16_t* p = tile + (lane & 15) * rowstride + ((lane >> 4) << 3);
  v8bf lo = *(const v8bf*)(p);
  v8bf hi = *(const v8bf*)(p + 16);
  v16bf r;
#pragma unroll
  for (int i = 0; i < 8; ++i) { r[i] = lo[i]; r[i + 8] = hi[i]; }
  return r;
}

// ---------------------------------------------------------------------------
// GEMM 1: qkv = x[8192x2048] @ w_qkv[2048x6144] (fp32 in, bf16 out, q/k/v split)
// Block tile 128x128, BK=64 (2 wmma K-steps/stage), 8 waves in 4x2.
// ---------------------------------------------------------------------------
__global__ __launch_bounds__(256) void qkv_gemm_kernel(
    const float* __restrict__ X, const float* __restrict__ W,
    bf16_t* __restrict__ Qo, bf16_t* __restrict__ Ko, bf16_t* __restrict__ Vo) {
  const int K = 2048, N = 6144;
  __shared__ bf16_t ldsA[128 * 72];   // [128 m][64 k], stride 72 (16B-aligned rows)
  __shared__ bf16_t ldsBT[128 * 72];  // [128 n][64 k] = B^T

  const int tid = threadIdx.x;
  const int wave = tid >> 5, lane = tid & 31;
  const int half = lane >> 4, ln = lane & 15;
  const int waveM = wave >> 1, waveN = wave & 1;
  const int n0 = blockIdx.x * 128, m0 = blockIdx.y * 128;

  v8f acc[2][4];
  const v8f vzero = {0.f, 0.f, 0.f, 0.f, 0.f, 0.f, 0.f, 0.f};
#pragma unroll
  for (int si = 0; si < 2; ++si)
#pragma unroll
    for (int ni = 0; ni < 4; ++ni) acc[si][ni] = vzero;

  for (int k0 = 0; k0 < K; k0 += 64) {
    // A tile: float4 loads, convert, packed b64 LDS stores
#pragma unroll
    for (int i = 0; i < 8; ++i) {
      int s = i * 256 + tid;
      int r = s >> 4, c4 = (s & 15) << 2;
      float4 f = *(const float4*)&X[(size_t)(m0 + r) * K + (k0 + c4)];
      v4bf b = {(bf16_t)f.x, (bf16_t)f.y, (bf16_t)f.z, (bf16_t)f.w};
      *(v4bf*)(&ldsA[r * 72 + c4]) = b;
    }
    // B tile transposed: float4 loads over n, scalar b16 scatter
#pragma unroll
    for (int i = 0; i < 8; ++i) {
      int s = i * 256 + tid;
      int kk = s >> 5, n4 = (s & 31) << 2;
      float4 f = *(const float4*)&W[(size_t)(k0 + kk) * N + (n0 + n4)];
      ldsBT[(n4 + 0) * 72 + kk] = (bf16_t)f.x;
      ldsBT[(n4 + 1) * 72 + kk] = (bf16_t)f.y;
      ldsBT[(n4 + 2) * 72 + kk] = (bf16_t)f.z;
      ldsBT[(n4 + 3) * 72 + kk] = (bf16_t)f.w;
    }
    __syncthreads();
#pragma unroll
    for (int kc = 0; kc < 2; ++kc) {
      v16bf a[2], b[4];
#pragma unroll
      for (int si = 0; si < 2; ++si)
        a[si] = load_frag16x32(ldsA + (waveM * 32 + si * 16) * 72 + kc * 32, 72);
#pragma unroll
      for (int ni = 0; ni < 4; ++ni)
        b[ni] = load_frag16x32(ldsBT + (waveN * 64 + ni * 16) * 72 + kc * 32, 72);
#pragma unroll
      for (int si = 0; si < 2; ++si)
#pragma unroll
        for (int ni = 0; ni < 4; ++ni)
          acc[si][ni] = WMMA_BF16(a[si], b[ni], acc[si][ni]);
    }
    __syncthreads();
  }

  const int which = n0 >> 11;            // 128-wide tile entirely in q, k, or v
  bf16_t* dst = (which == 0) ? Qo : (which == 1) ? Ko : Vo;
  const int e0 = n0 & 2047;
#pragma unroll
  for (int si = 0; si < 2; ++si)
#pragma unroll
    for (int ni = 0; ni < 4; ++ni)
#pragma unroll
      for (int j = 0; j < 8; ++j) {
        int row = m0 + waveM * 32 + si * 16 + j + 8 * half;
        int col = e0 + waveN * 64 + ni * 16 + ln;
        dst[(size_t)row * 2048 + col] = (bf16_t)acc[si][ni][j];
      }
}

// ---------------------------------------------------------------------------
// Flash attention. Grid: B*H*(S/128)=1024 WGs; each wave owns 16 query rows.
// Head slabs are contiguous (raw-view reshape is a no-op): base + b*S*E + h*S*D.
// ---------------------------------------------------------------------------
__global__ __launch_bounds__(256) void attn_kernel(
    const bf16_t* __restrict__ Qb, const bf16_t* __restrict__ Kb,
    const bf16_t* __restrict__ Vb, bf16_t* __restrict__ Ob) {
  const int D = 128;
  __shared__ bf16_t ldsK[64 * 136];      // [64 k][128 d], stride 136
  __shared__ bf16_t ldsVT[128 * 72];     // [128 d][64 k] = V^T, stride 72
  __shared__ bf16_t ldsP[8 * 16 * 72];   // per-wave [16 q][64 k], stride 72

  const int tid = threadIdx.x, wave = tid >> 5, lane = tid & 31;
  const int half = lane >> 4, ln = lane & 15;
  const int wg = blockIdx.x;
  const int qblk = wg & 15, h = (wg >> 4) & 15, b = wg >> 8;

  const size_t head = (size_t)b * 4194304 + (size_t)h * 262144;  // b*S*E + h*S*D
  const bf16_t* Qh = Qb + head;
  const bf16_t* Kh = Kb + head;
  const bf16_t* Vh = Vb + head;
  bf16_t* Oh = Ob + head;
  const int qr0 = qblk * 128 + wave * 16;

  // Q fragments: 16 rows x 128 d = 4 A-frags, straight from global
  v16bf aq[4];
#pragma unroll
  for (int kc = 0; kc < 4; ++kc) {
    const bf16_t* p = Qh + (size_t)(qr0 + ln) * D + kc * 32 + (half << 3);
    v8bf lo = *(const v8bf*)(p);
    v8bf hi = *(const v8bf*)(p + 16);
#pragma unroll
    for (int i = 0; i < 8; ++i) { aq[kc][i] = lo[i]; aq[kc][i + 8] = hi[i]; }
  }

  const v8f vzero = {0.f, 0.f, 0.f, 0.f, 0.f, 0.f, 0.f, 0.f};
  v8f accO[8];
#pragma unroll
  for (int dc = 0; dc < 8; ++dc) accO[dc] = vzero;
  float m[8], l[8];
#pragma unroll
  for (int j = 0; j < 8; ++j) { m[j] = -3.0e38f; l[j] = 0.f; }

  bf16_t* ldsPw = ldsP + wave * 16 * 72;
  const float scale = 0.08838834764831845f;  // 1/sqrt(128)
  const int nkt = qblk * 2 + 2;              // causal: tiles up to the diagonal

  for (int kt = 0; kt < nkt; ++kt) {
    const int kb0 = kt * 64;
    // stage K tile row-major: async global->LDS b128 copies
#pragma unroll
    for (int i = 0; i < 4; ++i) {
      int s = i * 256 + tid;
      int r = s >> 4, cb = (s & 15) << 3;
      copy16_g2l(Kh + (size_t)(kb0 + r) * D + cb, &ldsK[r * 136 + cb]);
    }
    // stage V transposed: vector b128 global reads, b16 LDS scatter
#pragma unroll
    for (int i = 0; i < 4; ++i) {
      int s = i * 256 + tid;
      int kk = s >> 4, db = (s & 15) << 3;
      v8bf v = *(const v8bf*)(Vh + (size_t)(kb0 + kk) * D + db);
#pragma unroll
      for (int e = 0; e < 8; ++e) ldsVT[(db + e) * 72 + kk] = v[e];
    }
    if (kt + 1 < nkt) {   // prefetch next tiles (global_prefetch_b8)
      __builtin_prefetch(Kh + (size_t)(kb0 + 64) * D + ln * 16, 0, 1);
      __builtin_prefetch(Vh + (size_t)(kb0 + 64) * D + ln * 16, 0, 1);
    }
    async_stage_wait();
    __syncthreads();

    // S = Q K^T : 4 k-col blocks x 4 d-chunks
    v8f sc[4];
#pragma unroll
    for (int t = 0; t < 4; ++t) {
      sc[t] = vzero;
#pragma unroll
      for (int kc = 0; kc < 4; ++kc) {
        v16bf bk = load_frag16x32(ldsK + (t * 16) * 136 + kc * 32, 136);
        sc[t] = WMMA_BF16(aq[kc], bk, sc[t]);
      }
    }

    // scale + causal additive mask, tile row-max
    float tmax[8];
#pragma unroll
    for (int j = 0; j < 8; ++j) tmax[j] = -3.0e38f;
#pragma unroll
    for (int t = 0; t < 4; ++t) {
      int kcol = kb0 + t * 16 + ln;
#pragma unroll
      for (int j = 0; j < 8; ++j) {
        int qrow = qr0 + j + 8 * half;
        float vv = sc[t][j] * scale + ((kcol <= qrow) ? 1.0f : -10000.0f);
        sc[t][j] = vv;
        tmax[j] = fmaxf(tmax[j], vv);
      }
    }
#pragma unroll
    for (int off = 1; off < 16; off <<= 1)
#pragma unroll
      for (int j = 0; j < 8; ++j)
        tmax[j] = fmaxf(tmax[j], __shfl_xor(tmax[j], off, 32));

    // online softmax update
    float alpha[8], rs[8];
#pragma unroll
    for (int j = 0; j < 8; ++j) {
      float mnew = fmaxf(m[j], tmax[j]);
      alpha[j] = __expf(m[j] - mnew);
      m[j] = mnew;
      rs[j] = 0.f;
    }
#pragma unroll
    for (int t = 0; t < 4; ++t)
#pragma unroll
      for (int j = 0; j < 8; ++j) {
        float p = __expf(sc[t][j] - m[j]);
        sc[t][j] = p;
        rs[j] += p;
      }
#pragma unroll
    for (int off = 1; off < 16; off <<= 1)
#pragma unroll
      for (int j = 0; j < 8; ++j) rs[j] += __shfl_xor(rs[j], off, 32);
#pragma unroll
    for (int j = 0; j < 8; ++j) l[j] = l[j] * alpha[j] + rs[j];
#pragma unroll
    for (int dc = 0; dc < 8; ++dc)
#pragma unroll
      for (int j = 0; j < 8; ++j) accO[dc][j] *= alpha[j];

    // C-layout P -> wave-private LDS (A-layout staging)
#pragma unroll
    for (int t = 0; t < 4; ++t)
#pragma unroll
      for (int j = 0; j < 8; ++j)
        ldsPw[(j + 8 * half) * 72 + t * 16 + ln] = (bf16_t)sc[t][j];

    // O += P V  (K-dim 64 = 2 wmma steps, 8 d-chunks)
    v16bf ap[2];
    ap[0] = load_frag16x32(ldsPw, 72);
    ap[1] = load_frag16x32(ldsPw + 32, 72);
#pragma unroll
    for (int dc = 0; dc < 8; ++dc)
#pragma unroll
      for (int k2 = 0; k2 < 2; ++k2) {
        v16bf bv = load_frag16x32(ldsVT + (dc * 16) * 72 + k2 * 32, 72);
        accO[dc] = WMMA_BF16(ap[k2], bv, accO[dc]);
      }
    __syncthreads();
  }

  // normalize and store (inverse raw-view reshape is identity on this layout)
#pragma unroll
  for (int j = 0; j < 8; ++j) {
    float inv = 1.0f / l[j];
#pragma unroll
    for (int dc = 0; dc < 8; ++dc)
      Oh[(size_t)(qr0 + j + 8 * half) * D + dc * 16 + ln] = (bf16_t)(accO[dc][j] * inv);
  }
}

// ---------------------------------------------------------------------------
// GEMM 3: out = o[8192x2048](bf16) @ w_out[2048x2048](fp32->bf16), fp32 out
// A tile staged with async global->LDS b128 copies (already bf16).
// ---------------------------------------------------------------------------
__global__ __launch_bounds__(256) void out_gemm_kernel(
    const bf16_t* __restrict__ A, const float* __restrict__ W,
    float* __restrict__ Out) {
  const int K = 2048, N = 2048;
  __shared__ bf16_t ldsA[128 * 72];
  __shared__ bf16_t ldsBT[128 * 72];

  const int tid = threadIdx.x;
  const int wave = tid >> 5, lane = tid & 31;
  const int half = lane >> 4, ln = lane & 15;
  const int waveM = wave >> 1, waveN = wave & 1;
  const int n0 = blockIdx.x * 128, m0 = blockIdx.y * 128;

  v8f acc[2][4];
  const v8f vzero = {0.f, 0.f, 0.f, 0.f, 0.f, 0.f, 0.f, 0.f};
#pragma unroll
  for (int si = 0; si < 2; ++si)
#pragma unroll
    for (int ni = 0; ni < 4; ++ni) acc[si][ni] = vzero;

  for (int k0 = 0; k0 < K; k0 += 64) {
    // A tile: 128 rows x 64 bf16 = 1024 x 16B slots, async copies
#pragma unroll
    for (int i = 0; i < 4; ++i) {
      int s = i * 256 + tid;
      int r = s >> 3, cb = (s & 7) << 3;
      copy16_g2l(A + (size_t)(m0 + r) * K + (k0 + cb), &ldsA[r * 72 + cb]);
    }
    // B tile transposed: fp32 -> bf16
#pragma unroll
    for (int i = 0; i < 8; ++i) {
      int s = i * 256 + tid;
      int kk = s >> 5, n4 = (s & 31) << 2;
      float4 f = *(const float4*)&W[(size_t)(k0 + kk) * N + (n0 + n4)];
      ldsBT[(n4 + 0) * 72 + kk] = (bf16_t)f.x;
      ldsBT[(n4 + 1) * 72 + kk] = (bf16_t)f.y;
      ldsBT[(n4 + 2) * 72 + kk] = (bf16_t)f.z;
      ldsBT[(n4 + 3) * 72 + kk] = (bf16_t)f.w;
    }
    async_stage_wait();
    __syncthreads();
#pragma unroll
    for (int kc = 0; kc < 2; ++kc) {
      v16bf a[2], b[4];
#pragma unroll
      for (int si = 0; si < 2; ++si)
        a[si] = load_frag16x32(ldsA + (waveM * 32 + si * 16) * 72 + kc * 32, 72);
#pragma unroll
      for (int ni = 0; ni < 4; ++ni)
        b[ni] = load_frag16x32(ldsBT + (waveN * 64 + ni * 16) * 72 + kc * 32, 72);
#pragma unroll
      for (int si = 0; si < 2; ++si)
#pragma unroll
        for (int ni = 0; ni < 4; ++ni)
          acc[si][ni] = WMMA_BF16(a[si], b[ni], acc[si][ni]);
    }
    __syncthreads();
  }

#pragma unroll
  for (int si = 0; si < 2; ++si)
#pragma unroll
    for (int ni = 0; ni < 4; ++ni)
#pragma unroll
      for (int j = 0; j < 8; ++j) {
        int row = m0 + waveM * 32 + si * 16 + j + 8 * half;
        int col = n0 + waveN * 64 + ni * 16 + ln;
        Out[(size_t)row * 2048 + col] = acc[si][ni][j];
      }
}

// ---------------------------------------------------------------------------
extern "C" void kernel_launch(void* const* d_in, const int* in_sizes, int n_in,
                              void* d_out, int out_size, void* d_ws, size_t ws_size,
                              hipStream_t stream) {
  const float* x     = (const float*)d_in[0];  // [4,2048,2048]
  const float* w_qkv = (const float*)d_in[1];  // [2048,6144]
  const float* w_out = (const float*)d_in[2];  // [2048,2048]
  float* out = (float*)d_out;                  // [4,2048,2048]

  const size_t elems = (size_t)4 * 2048 * 2048;  // per-buffer bf16 elements
  bf16_t* qb = (bf16_t*)d_ws;
  bf16_t* kb = qb + elems;
  bf16_t* vb = kb + elems;
  bf16_t* ob = vb + elems;   // total scratch: 4 * 32 MiB = 128 MiB

  qkv_gemm_kernel<<<dim3(48, 64), 256, 0, stream>>>(x, w_qkv, qb, kb, vb);
  attn_kernel<<<dim3(1024), 256, 0, stream>>>(qb, kb, vb, ob);
  out_gemm_kernel<<<dim3(16, 64), 256, 0, stream>>>(ob, w_out, out);
}
